// MHAx2Test_11871289606773
// MI455X (gfx1250) — matrix operand order repaired and checked by reading, not compiled, not run on hardware
//
#include <hip/hip_runtime.h>
#include <hip/hip_bf16.h>

typedef __bf16 bf16;
typedef __attribute__((ext_vector_type(16))) __bf16 v16bf;
typedef __attribute__((ext_vector_type(8)))  float  v8f;

struct __align__(16) U4 { unsigned int x, y, z, w; };
struct __align__(16) F4 { float x, y, z, w; };
union Frag { U4 u[2]; v16bf v; };

static __device__ __forceinline__ v8f zero8() {
  v8f z;
#pragma unroll
  for (int i = 0; i < 8; ++i) z[i] = 0.0f;
  return z;
}

static __device__ __forceinline__ v8f wmma_bf16(const Frag& a, const Frag& b, v8f c) {
  return __builtin_amdgcn_wmma_f32_16x16x32_bf16(false, a.v, false, b.v,
                                                 (short)0, c, false, false);
}

// CDNA5 async copy: 16B global -> LDS, tracked by ASYNCcnt (ISA 08, opcode 98).
static __device__ __forceinline__ void async_copy16(unsigned lds_off, const bf16* g) {
  unsigned long long ga = (unsigned long long)(uintptr_t)g;
  asm volatile("global_load_async_to_lds_b128 %0, %1, off"
               :: "v"(lds_off), "v"(ga) : "memory");
}
static __device__ __forceinline__ void wait_async0() {
  asm volatile("s_wait_asynccnt 0x0" ::: "memory");
}
static __device__ __forceinline__ unsigned lds_off_of(const void* p) {
  return (unsigned)(uintptr_t)p;  // generic LDS addr: low 32 bits are the LDS offset
}

// ---------------------------------------------------------------------------
// Utility kernels
// ---------------------------------------------------------------------------
__global__ void cvt_kernel(const float* __restrict__ src, bf16* __restrict__ dst, int n) {
  int i = blockIdx.x * 256 + threadIdx.x;
  if (i < n) dst[i] = (bf16)src[i];
}

// ---------------------------------------------------------------------------
// conv1: x[4,3,128,512], w[32,3,k,3], pad_w=1 over the 512 dim.
// out: h[4,32,W,512] (f32 + bf16 shadow). Memory-bound; scalar is fine.
// ---------------------------------------------------------------------------
__global__ __launch_bounds__(256) void conv1_kernel(
    const float* __restrict__ x, const float* __restrict__ wgt,
    const float* __restrict__ bias, float* __restrict__ hf,
    bf16* __restrict__ hb, int W, int kk) {
  int idx = blockIdx.x * 256 + threadIdx.x;
  int d = idx & 511;
  int t = idx >> 9;
  int wpos = t % W; t /= W;
  int f = t & 31;
  int b = t >> 5;
  float acc = bias[f];
  for (int c = 0; c < 3; ++c) {
    for (int kh = 0; kh < kk; ++kh) {
      const float* xp = x + (((size_t)(b * 3 + c) * 128 + (wpos + kh)) << 9);
      const float* wp = wgt + ((size_t)(f * 3 + c) * kk + kh) * 3;
      if (d >= 1)   acc += xp[d - 1] * wp[0];
      acc += xp[d] * wp[1];
      if (d < 511)  acc += xp[d + 1] * wp[2];
    }
  }
  hf[idx] = acc;
  hb[idx] = (bf16)acc;
}

// ---------------------------------------------------------------------------
// GEMM: C[M,N] = A[M,K](bf16) @ Wt[N,K](bf16)^T + bias + resid
// 128x128 tile / 8 waves, 32x64 per wave (8 WMMA accums).
// Double-buffered LDS, tiles filled by global_load_async_to_lds_b128:
// issue stage n+1, WMMA on stage n, s_wait_asynccnt + one barrier per step.
// ---------------------------------------------------------------------------
__global__ __launch_bounds__(256) void gemm_kernel(
    const bf16* __restrict__ A, const bf16* __restrict__ Bw,
    const float* __restrict__ bias, const float* __restrict__ resid,
    float* __restrict__ Cf, bf16* __restrict__ Cb, int M, int N, int K) {
  constexpr int LDK = 40;                  // padded bf16 row stride (80B, 16B-aligned)
  __shared__ bf16 lA[2][128 * LDK];
  __shared__ bf16 lB[2][128 * LDK];
  const int m0 = blockIdx.x * 128, n0 = blockIdx.y * 128;
  const int t = threadIdx.x, wave = t >> 5, lane = t & 31;
  const int wm = (wave >> 1) * 32, wn = (wave & 1) * 64;
  const int lrow = lane & 15;
  const int koffA = (lane >> 4) << 3;      // A frag: 8-elem chunks at {k, k+16}
  const int koffB = (lane >> 4) << 3;      // B frag: 8-elem chunks at {k, k+16}
  v8f acc[2][4];
#pragma unroll
  for (int i = 0; i < 2; ++i)
#pragma unroll
    for (int j = 0; j < 4; ++j) acc[i][j] = zero8();

  const int rS = t >> 1, cS = (t & 1) * 16;
  const bf16* gA = A + (size_t)(m0 + rS) * K + cS;
  const bf16* gB = Bw + (size_t)(n0 + rS) * K + cS;
  unsigned aDst[2], bDst[2];
#pragma unroll
  for (int sgi = 0; sgi < 2; ++sgi) {
    aDst[sgi] = lds_off_of(&lA[sgi][rS * LDK + cS]);
    bDst[sgi] = lds_off_of(&lB[sgi][rS * LDK + cS]);
  }
  // prologue: stage 0 <- kt 0
  async_copy16(aDst[0],      gA);
  async_copy16(aDst[0] + 16, gA + 8);
  async_copy16(bDst[0],      gB);
  async_copy16(bDst[0] + 16, gB + 8);
  wait_async0();
  __syncthreads();

  for (int kt = 0; kt < K; kt += 32) {
    const int cur = (kt >> 5) & 1, nxt = cur ^ 1;
    if (kt + 32 < K) {  // prefetch next K-tile into the other stage
      const bf16* ga = gA + kt + 32;
      const bf16* gb = gB + kt + 32;
      async_copy16(aDst[nxt],      ga);
      async_copy16(aDst[nxt] + 16, ga + 8);
      async_copy16(bDst[nxt],      gb);
      async_copy16(bDst[nxt] + 16, gb + 8);
    }
    Frag fa[2], fb[4];
#pragma unroll
    for (int i = 0; i < 2; ++i) {
      const bf16* p = &lA[cur][(wm + i * 16 + lrow) * LDK + koffA];
      fa[i].u[0] = *(const U4*)p;
      fa[i].u[1] = *(const U4*)(p + 16);
    }
#pragma unroll
    for (int j = 0; j < 4; ++j) {
      const bf16* p = &lB[cur][(wn + j * 16 + lrow) * LDK + koffB];
      fb[j].u[0] = *(const U4*)p;
      fb[j].u[1] = *(const U4*)(p + 16);
    }
#pragma unroll
    for (int i = 0; i < 2; ++i)
#pragma unroll
      for (int j = 0; j < 4; ++j) acc[i][j] = wmma_bf16(fa[i], fb[j], acc[i][j]);
    wait_async0();
    __syncthreads();
  }

  const int rowHalf = (lane >> 4) << 3;
#pragma unroll
  for (int i = 0; i < 2; ++i)
#pragma unroll
    for (int j = 0; j < 4; ++j)
#pragma unroll
      for (int r = 0; r < 8; ++r) {
        int row = m0 + wm + i * 16 + rowHalf + r;
        int col = n0 + wn + j * 16 + lrow;
        float v = acc[i][j][r];
        if (bias)  v += bias[col];
        if (resid) v += resid[(size_t)row * N + col];
        if (Cf) Cf[(size_t)row * N + col] = v;
        if (Cb) Cb[(size_t)row * N + col] = (bf16)v;
      }
}

// ---------------------------------------------------------------------------
// Attention: one WG per (seq s in [0,128), head h). W <= 128, dh = 64.
// S = QK^T/8 via WMMA, register softmax (shfl across 16-lane halves),
// P bounced via LDS (accumulator->A-layout transpose), O = P V via WMMA.
// ---------------------------------------------------------------------------
__global__ __launch_bounds__(256) void attn_kernel(
    const bf16* __restrict__ Q, const bf16* __restrict__ Km,
    const bf16* __restrict__ V, bf16* __restrict__ O, int W) {
  __shared__ bf16 lK[128 * 64];    // keys   [u][d]      16 KB
  __shared__ bf16 lVT[64 * 128];   // V^T    [d][u]      16 KB
  __shared__ bf16 lP[128 * 128];   // probs  [w][u]      32 KB
  const int s = blockIdx.x, h = blockIdx.y;
  const int col0 = h * 64;
  const size_t base = (size_t)s * W;
  const int t = threadIdx.x, wave = t >> 5, lane = t & 31;
  const int lrow = lane & 15;
  const int koffA = (lane >> 4) << 3;
  const int koffB = (lane >> 4) << 3;

  {  // async-stage K tile (rows beyond W read the next sequence: masked later)
    int u = t >> 1, c = (t & 1) * 32;
    const bf16* g = Km + (base + u) * 512 + col0 + c;
    unsigned dst = lds_off_of(&lK[u * 64 + c]);
    async_copy16(dst,      g);
    async_copy16(dst + 16, g + 8);
    async_copy16(dst + 32, g + 16);
    async_copy16(dst + 48, g + 24);
  }
  {  // load V transposed (gather -> plain LDS stores)
    int d = t & 63, u0 = (t >> 6) * 32;
    for (int u = 0; u < 32; ++u)
      lVT[d * 128 + u0 + u] = V[(base + u0 + u) * 512 + col0 + d];
  }
  wait_async0();
  __syncthreads();

  const int wq0 = wave * 16;
  Frag fq[2];
#pragma unroll
  for (int c = 0; c < 2; ++c) {
    const bf16* p = Q + (base + wq0 + lrow) * 512 + col0 + c * 32 + koffA;
    fq[c].u[0] = *(const U4*)p;
    fq[c].u[1] = *(const U4*)(p + 16);
  }
  v8f sf[8];
#pragma unroll
  for (int j = 0; j < 8; ++j) sf[j] = zero8();
#pragma unroll
  for (int j = 0; j < 8; ++j) {
#pragma unroll
    for (int c = 0; c < 2; ++c) {
      Frag fk;
      const bf16* p = &lK[(j * 16 + lrow) * 64 + c * 32 + koffB];
      fk.u[0] = *(const U4*)p;
      fk.u[1] = *(const U4*)(p + 16);
      sf[j] = wmma_bf16(fq[c], fk, sf[j]);
    }
  }

  // softmax: row (r + 8*(lane>=16)) lives across the 16 lanes of a half
  const float scale = 0.125f;
  const int rowHalf = (lane >> 4) << 3;
#pragma unroll
  for (int r = 0; r < 8; ++r) {
    float vals[8];
    float m = -1e30f;
#pragma unroll
    for (int j = 0; j < 8; ++j) {
      float xv = sf[j][r] * scale;
      if (j * 16 + lrow >= W) xv = -1e30f;
      vals[j] = xv;
      m = fmaxf(m, xv);
    }
    for (int d2 = 1; d2 < 16; d2 <<= 1) m = fmaxf(m, __shfl_xor(m, d2, 16));
    float sum = 0.0f;
#pragma unroll
    for (int j = 0; j < 8; ++j) {
      float e = __expf(vals[j] - m);
      if (j * 16 + lrow >= W) e = 0.0f;
      vals[j] = e;
      sum += e;
    }
    for (int d2 = 1; d2 < 16; d2 <<= 1) sum += __shfl_xor(sum, d2, 16);
    float inv = 1.0f / sum;
    int rl = wq0 + rowHalf + r;
#pragma unroll
    for (int j = 0; j < 8; ++j)
      lP[rl * 128 + j * 16 + lrow] = (bf16)(vals[j] * inv);
  }
  __syncthreads();

  // O = P @ V  (K dim = 128 padded keys; masked cols were zeroed above)
  v8f of[4];
#pragma unroll
  for (int j = 0; j < 4; ++j) of[j] = zero8();
#pragma unroll
  for (int c = 0; c < 4; ++c) {
    Frag fp;
    const bf16* p = &lP[(wq0 + lrow) * 128 + c * 32 + koffA];
    fp.u[0] = *(const U4*)p;
    fp.u[1] = *(const U4*)(p + 16);
#pragma unroll
    for (int j = 0; j < 4; ++j) {
      Frag fv;
      const bf16* pv = &lVT[(j * 16 + lrow) * 128 + c * 32 + koffB];
      fv.u[0] = *(const U4*)pv;
      fv.u[1] = *(const U4*)(pv + 16);
      of[j] = wmma_bf16(fp, fv, of[j]);
    }
  }
#pragma unroll
  for (int j = 0; j < 4; ++j)
#pragma unroll
    for (int r = 0; r < 8; ++r) {
      int rl = wq0 + rowHalf + r;
      if (rl < W)  // don't clobber the next sequence's tokens
        O[(base + rl) * 512 + col0 + j * 16 + lrow] = (bf16)of[j][r];
    }
}

// ---------------------------------------------------------------------------
// conv2 as WMMA GEMM with K split across gridDim.y = 8 (4 input channels per
// WG). Each WG stores its own disjoint partial slice (deterministic, no float
// atomics). rows = (b, w'), cols = 32 filters, K-slice = 4*k*512.
// ---------------------------------------------------------------------------
__global__ __launch_bounds__(256) void conv2_kernel(
    const float* __restrict__ H, const float* __restrict__ W2,
    float* __restrict__ partial, int Wlen, int kk) {
  constexpr int LDK = 40;
  __shared__ bf16 lA[128 * LDK];
  __shared__ bf16 lB[32 * LDK];
  const int Wp = Wlen - kk + 1;
  const int Mrows = 4 * Wp;
  const int K = 32 * kk * 512;          // full K (weight row stride)
  const int Kslice = 4 * kk * 512;      // this WG's K extent
  const int ks = blockIdx.y;            // channel-group 0..7
  const int kbase = ks * Kslice;
  const int m0 = blockIdx.x * 128;
  const int t = threadIdx.x, wave = t >> 5, lane = t & 31;
  const int lrow = lane & 15;
  const int koffA = (lane >> 4) << 3;
  const int koffB = (lane >> 4) << 3;
  v8f acc[2];
  acc[0] = zero8();
  acc[1] = zero8();
  const int rS = t >> 1, cS = (t & 1) * 16;

  for (int kt = 0; kt < Kslice; kt += 32) {
    {  // stage A: 16 contiguous f32 -> packed bf16 (32-runs never cross (ci,dh))
      int row = m0 + rS;
      int j = kt + cS;
      int d0 = j & 511;
      int q = j >> 9;
      int dh = q % kk, ci = ks * 4 + q / kk;
      if (row < Mrows) {
        int b = row / Wp, wp = row - b * Wp;
        const F4* s4 = (const F4*)(H + (((size_t)(b * 32 + ci) * Wlen + wp + dh) << 9) + d0);
        float fv[16];
#pragma unroll
        for (int q4 = 0; q4 < 4; ++q4) {
          F4 vv = s4[q4];
          fv[4 * q4] = vv.x; fv[4 * q4 + 1] = vv.y;
          fv[4 * q4 + 2] = vv.z; fv[4 * q4 + 3] = vv.w;
        }
#pragma unroll
        for (int e = 0; e < 16; e += 2) {
          unsigned lo = (unsigned)__builtin_bit_cast(unsigned short, (bf16)fv[e]);
          unsigned hi = (unsigned)__builtin_bit_cast(unsigned short, (bf16)fv[e + 1]);
          *(unsigned*)&lA[rS * LDK + cS + e] = lo | (hi << 16);
        }
      } else {
#pragma unroll
        for (int e = 0; e < 16; e += 2) *(unsigned*)&lA[rS * LDK + cS + e] = 0u;
      }
    }
    if (t < 64) {  // stage weights (shared by all waves)
      int f = t >> 1, c0 = (t & 1) * 16;
      const F4* s4 = (const F4*)(W2 + (size_t)f * K + kbase + kt + c0);
      float fv[16];
#pragma unroll
      for (int q4 = 0; q4 < 4; ++q4) {
        F4 vv = s4[q4];
        fv[4 * q4] = vv.x; fv[4 * q4 + 1] = vv.y;
        fv[4 * q4 + 2] = vv.z; fv[4 * q4 + 3] = vv.w;
      }
#pragma unroll
      for (int e = 0; e < 16; e += 2) {
        unsigned lo = (unsigned)__builtin_bit_cast(unsigned short, (bf16)fv[e]);
        unsigned hi = (unsigned)__builtin_bit_cast(unsigned short, (bf16)fv[e + 1]);
        *(unsigned*)&lB[f * LDK + c0 + e] = lo | (hi << 16);
      }
    }
    __syncthreads();
    Frag fa;
    const bf16* pa = &lA[(wave * 16 + lrow) * LDK + koffA];
    fa.u[0] = *(const U4*)pa;
    fa.u[1] = *(const U4*)(pa + 16);
#pragma unroll
    for (int j = 0; j < 2; ++j) {
      Frag fb;
      const bf16* pb = &lB[(j * 16 + lrow) * LDK + koffB];
      fb.u[0] = *(const U4*)pb;
      fb.u[1] = *(const U4*)(pb + 16);
      acc[j] = wmma_bf16(fa, fb, acc[j]);
    }
    __syncthreads();
  }

  const int rowHalf = (lane >> 4) << 3;
#pragma unroll
  for (int j = 0; j < 2; ++j)
#pragma unroll
    for (int r = 0; r < 8; ++r) {
      int row = m0 + wave * 16 + rowHalf + r;
      if (row < Mrows) {
        int f = j * 16 + lrow;
        partial[((size_t)ks * 512 + row) * 32 + f] = acc[j][r];
      }
    }
}

// Sum the 8 K-slices, add bias, relu, max over positions. One thread per (b,f).
__global__ void conv2_finalize(const float* __restrict__ partial,
                               const float* __restrict__ bias,
                               float* __restrict__ hidden, int Wp, int fi) {
  int t = threadIdx.x;
  if (t < 128) {
    int b = t >> 5, f = t & 31;
    float mx = 0.0f;  // relu >= 0
    for (int wp = 0; wp < Wp; ++wp) {
      int row = b * Wp + wp;
      float sv = bias[f];
#pragma unroll
      for (int ks = 0; ks < 8; ++ks) sv += partial[((size_t)ks * 512 + row) * 32 + f];
      mx = fmaxf(mx, fmaxf(sv, 0.0f));
    }
    hidden[b * 320 + fi * 32 + f] = mx;
  }
}

// ---------------------------------------------------------------------------
// Final FC: out[4,2] = hidden[4,320] @ fc_w[2,320]^T + fc_b  (tiny)
// ---------------------------------------------------------------------------
__global__ void fc_kernel(const float* __restrict__ hidden,
                          const float* __restrict__ w,
                          const float* __restrict__ b, float* __restrict__ out) {
  int t = threadIdx.x;
  if (t < 8) {
    int bb = t >> 1, j = t & 1;
    float acc = b[j];
    for (int c = 0; c < 320; ++c) acc += hidden[bb * 320 + c] * w[j * 320 + c];
    out[t] = acc;
  }
}

// ---------------------------------------------------------------------------
extern "C" void kernel_launch(void* const* d_in, const int* in_sizes, int n_in,
                              void* d_out, int out_size, void* d_ws, size_t ws_size,
                              hipStream_t stream) {
  (void)in_sizes; (void)n_in; (void)out_size; (void)ws_size;
  const float* x = (const float*)d_in[0];
  const float *c1w[10], *c1b[10], *c2w[10], *c2b[10];
  for (int i = 0; i < 10; ++i) {
    c1w[i] = (const float*)d_in[1 + i];
    c1b[i] = (const float*)d_in[11 + i];
    c2w[i] = (const float*)d_in[21 + i];
    c2b[i] = (const float*)d_in[31 + i];
  }
  const float *Wq[2], *Bq[2], *Wk[2], *Bk[2], *Wv[2], *Bv[2], *Wo[2], *Bo[2];
  for (int p = 0; p < 2; ++p) {
    int o = 41 + p * 8;
    Wq[p] = (const float*)d_in[o + 0]; Bq[p] = (const float*)d_in[o + 1];
    Wk[p] = (const float*)d_in[o + 2]; Bk[p] = (const float*)d_in[o + 3];
    Wv[p] = (const float*)d_in[o + 4]; Bv[p] = (const float*)d_in[o + 5];
    Wo[p] = (const float*)d_in[o + 6]; Bo[p] = (const float*)d_in[o + 7];
  }
  const float* fcw = (const float*)d_in[57];
  const float* fcb = (const float*)d_in[58];

  // workspace carve-out
  unsigned char* p8 = (unsigned char*)d_ws;
  auto carve = [&](size_t bytes) {
    void* r = (void*)p8;
    p8 += (bytes + 255) & ~(size_t)255;
    return r;
  };
  const size_t ACT = (size_t)4 * 32 * 128 * 512;  // max activation elems
  float* hA  = (float*)carve(ACT * 4);
  float* hB  = (float*)carve(ACT * 4);
  bf16* hAb  = (bf16*)carve(ACT * 2);
  bf16* hBb  = (bf16*)carve(ACT * 2);
  bf16* qb   = (bf16*)carve(ACT * 2);
  bf16* kb   = (bf16*)carve(ACT * 2);
  bf16* vb   = (bf16*)carve(ACT * 2);
  bf16* ob   = (bf16*)carve(ACT * 2);
  float* part = (float*)carve((size_t)8 * 512 * 32 * 4);
  bf16* wmat[8];
  for (int m = 0; m < 8; ++m) wmat[m] = (bf16*)carve((size_t)512 * 512 * 2);

  float* out = (float*)d_out;
  float* hidden = out + 8;

  const float* wsrc[8] = {Wq[0], Wk[0], Wv[0], Wo[0], Wq[1], Wk[1], Wv[1], Wo[1]};
  for (int m = 0; m < 8; ++m)
    cvt_kernel<<<dim3(1024), 256, 0, stream>>>(wsrc[m], wmat[m], 512 * 512);

  for (int i = 0; i < 10; ++i) {
    int kk = i + 1, W = 129 - kk, M = 128 * W;  // M always multiple of 128
    conv1_kernel<<<dim3(256 * W), 256, 0, stream>>>(x, c1w[i], c1b[i], hA, hAb, W, kk);
    float* curF = hA; bf16* curB = hAb;
    float* nxtF = hB; bf16* nxtB = hBb;
    for (int blk = 0; blk < 2; ++blk) {
      int wb = blk * 4;
      gemm_kernel<<<dim3(W, 4), 256, 0, stream>>>(curB, wmat[wb + 0], Bq[blk],
                                                  nullptr, nullptr, qb, M, 512, 512);
      gemm_kernel<<<dim3(W, 4), 256, 0, stream>>>(curB, wmat[wb + 1], Bk[blk],
                                                  nullptr, nullptr, kb, M, 512, 512);
      gemm_kernel<<<dim3(W, 4), 256, 0, stream>>>(curB, wmat[wb + 2], Bv[blk],
                                                  nullptr, nullptr, vb, M, 512, 512);
      attn_kernel<<<dim3(128, 8), 256, 0, stream>>>(qb, kb, vb, ob, W);
      gemm_kernel<<<dim3(W, 4), 256, 0, stream>>>(ob, wmat[wb + 3], Bo[blk],
                                                  curF, nxtF, nxtB, M, 512, 512);
      { float* tf = curF; curF = nxtF; nxtF = tf;
        bf16* tb = curB; curB = nxtB; nxtB = tb; }
    }
    int Wp = W - kk + 1;
    conv2_kernel<<<dim3(4, 8), 256, 0, stream>>>(curF, c2w[i], part, W, kk);
    conv2_finalize<<<dim3(1), 128, 0, stream>>>(part, c2b[i], hidden, Wp, i);
  }
  fc_kernel<<<dim3(1), 64, 0, stream>>>(hidden, fcw, fcb, out);
}
